// Grace_auto_86998857548321
// MI455X (gfx1250) — compile-verified, matrix operand order checked by reading, not run on hardware
//
#include <hip/hip_runtime.h>
#include <hip/hip_bf16.h>

// ---------------------------------------------------------------------------
// Types for CDNA5 WMMA (wave32): v_wmma_f32_16x16x32_bf16
// ---------------------------------------------------------------------------
typedef __attribute__((ext_vector_type(16))) __bf16 v16bf;
typedef __attribute__((ext_vector_type(8)))  float  v8f;
typedef __attribute__((ext_vector_type(4)))  int    v4i;

// ---------------------------------------------------------------------------
// CDNA5 async global->LDS copy (16B per lane), tracked by ASYNCcnt.
// Builtin signature (from clang diagnostic): param0 = v4i in AS1 (global),
// param1 = v4i in LDS, then imm offset + cpol.
// ---------------------------------------------------------------------------
__device__ __forceinline__ void async_copy_b128(void* lds_dst, const void* gsrc) {
#if __has_builtin(__builtin_amdgcn_global_load_async_to_lds_b128)
    __attribute__((address_space(1))) v4i* g =
        (__attribute__((address_space(1))) v4i*)(__UINTPTR_TYPE__)gsrc;
    __attribute__((address_space(3))) v4i* l =
        (__attribute__((address_space(3))) v4i*)(unsigned)(__UINTPTR_TYPE__)lds_dst;
    __builtin_amdgcn_global_load_async_to_lds_b128(g, l, 0, 0);
#else
    unsigned loff = (unsigned)(__UINTPTR_TYPE__)lds_dst;   // LDS offset = low 32 bits
    unsigned long long ga = (unsigned long long)(__UINTPTR_TYPE__)gsrc;
    asm volatile("global_load_async_to_lds_b128 %0, %1, off"
                 :: "v"(loff), "v"(ga) : "memory");
#endif
}

__device__ __forceinline__ void wait_asynccnt0() {
#if __has_builtin(__builtin_amdgcn_s_wait_asynccnt)
    __builtin_amdgcn_s_wait_asynccnt(0);
#else
    asm volatile("s_wait_asynccnt 0x0" ::: "memory");
#endif
}

// Load a 16-element bf16 fragment: two 16B chunks (ISA 7.12.2 16-bit layout:
// lane kbase in {0,8}; VGPR0-3 hold K=kbase..kbase+7, VGPR4-7 K=kbase+16..+23).
__device__ __forceinline__ v16bf frag_load(const __bf16* base) {
    union { v16bf v; uint4 u[2]; } f;
    f.u[0] = *reinterpret_cast<const uint4*>(base);
    f.u[1] = *reinterpret_cast<const uint4*>(base + 16);
    return f.v;
}

// ---------------------------------------------------------------------------
// GEMM: H[n_rows x OUT] = X[n_rows x 128](bf16) @ Wt[OUT x 128](bf16, pre-T)
// Wt staged in LDS once per block via async copies; one wave per 16-row tile,
// A fragments for the whole K=128 in regs; 4x v_wmma per 16x16 output tile.
// ---------------------------------------------------------------------------
template<int OUT>
__global__ __launch_bounds__(256) void gemm_bf16_wmma(
    const __bf16* __restrict__ X, const __bf16* __restrict__ Wt,
    float* __restrict__ H, int n_rows)
{
    __shared__ __bf16 ldsw[OUT * 128];

    // Cooperative async stage of Wt (OUT*256 bytes) into LDS.
    constexpr int CHUNKS = OUT * 16;              // 16B chunks, multiple of 256
#pragma unroll
    for (int c = threadIdx.x; c < CHUNKS; c += 256)
        async_copy_b128((char*)ldsw + (size_t)c * 16,
                        (const char*)Wt + (size_t)c * 16);
    wait_asynccnt0();
    __syncthreads();

    const int wave = threadIdx.x >> 5;
    const int lane = threadIdx.x & 31;
    const int tile = blockIdx.x * 8 + wave;
    const int row0 = tile * 16;
    if (row0 >= n_rows) return;          // uniform per wave -> EXEC stays all-1s

    const int mn = lane & 15;            // M for A, N for B
    const int kb = (lane >> 4) << 3;     // 0 or 8

    v16bf a[4];
    const __bf16* xrow = X + (size_t)(row0 + mn) * 128;
#pragma unroll
    for (int ks = 0; ks < 4; ++ks)
        a[ks] = frag_load(xrow + ks * 32 + kb);

#pragma unroll
    for (int c = 0; c < OUT / 16; ++c) {
        const __bf16* wrow = ldsw + (size_t)(c * 16 + mn) * 128;   // ds_load path
        v8f acc = {};
#pragma unroll
        for (int ks = 0; ks < 4; ++ks) {
            v16bf b = frag_load(wrow + ks * 32 + kb);
            acc = __builtin_amdgcn_wmma_f32_16x16x32_bf16(
                false, a[ks], false, b, (short)0, acc, false, false);
        }
        // C layout: VGPR v -> row = row0 + v + 8*(lane>>4), col = c*16 + (lane&15)
        const int rbase = row0 + ((lane >> 4) << 3);
        const int col   = c * 16 + mn;
#pragma unroll
        for (int v = 0; v < 8; ++v)
            H[(size_t)(rbase + v) * OUT + col] = acc[v];
    }
}

// ---------------------------------------------------------------------------
// Graph preprocessing kernels
// ---------------------------------------------------------------------------
__global__ void deg_kernel(const long long* __restrict__ ei,
                           const float* __restrict__ ew,
                           long long E, float* __restrict__ deg)
{
    long long e = blockIdx.x * (long long)blockDim.x + threadIdx.x;
    if (e < E) unsafeAtomicAdd(&deg[ei[E + e]], ew[e]);
}

__global__ void rsqrt_kernel(float* __restrict__ d, long long n)
{
    long long i = blockIdx.x * (long long)blockDim.x + threadIdx.x;
    if (i < n) d[i] = rsqrtf(d[i] + 1.0f);
}

__global__ void norm_kernel(const long long* __restrict__ ei,
                            const float* __restrict__ ew,
                            const float* __restrict__ dinv,
                            long long E, float* __restrict__ nrm)
{
    long long e = blockIdx.x * (long long)blockDim.x + threadIdx.x;
    if (e < E) nrm[e] = dinv[ei[e]] * ew[e] * dinv[ei[E + e]];
}

// ---------------------------------------------------------------------------
// Conversions: x -> bf16; W[K x OUT] -> Wt[OUT x K] bf16 (transposed)
// ---------------------------------------------------------------------------
__global__ void conv_bf16_kernel(const float* __restrict__ in,
                                 __bf16* __restrict__ out, long long n)
{
    long long i = blockIdx.x * (long long)blockDim.x + threadIdx.x;
    if (i < n) out[i] = (__bf16)in[i];
}

__global__ void convw_kernel(const float* __restrict__ W,
                             __bf16* __restrict__ Wt, int K, int OUT)
{
    int i = blockIdx.x * blockDim.x + threadIdx.x;
    if (i < K * OUT) {
        int k = i / OUT, o = i % OUT;
        Wt[(size_t)o * K + k] = (__bf16)W[(size_t)k * OUT + o];
    }
}

// ---------------------------------------------------------------------------
// Edge scatter: AGG[dst] += norm[e] * H[src], float4-vectorized f32 atomics
// ---------------------------------------------------------------------------
template<int F>
__global__ void scatter_kernel(const float* __restrict__ H,
                               const long long* __restrict__ ei, long long E,
                               const float* __restrict__ nrm,
                               float* __restrict__ AGG)
{
    constexpr int G = F / 4;
    long long idx = blockIdx.x * (long long)blockDim.x + threadIdx.x;
    if (idx >= E * G) return;
    long long e = idx / G;
    int g = (int)(idx - e * G);
    long long s = ei[e], d = ei[E + e];
    float w = nrm[e];
    const float4 v = *reinterpret_cast<const float4*>(H + s * (size_t)F + g * 4);
    float* out = AGG + d * (size_t)F + g * 4;
    unsafeAtomicAdd(out + 0, v.x * w);
    unsafeAtomicAdd(out + 1, v.y * w);
    unsafeAtomicAdd(out + 2, v.z * w);
    unsafeAtomicAdd(out + 3, v.w * w);
}

// ---------------------------------------------------------------------------
// Finalize: out = relu(agg + h * dinv^2 + bias); bf16 variant feeds next GEMM
// ---------------------------------------------------------------------------
template<int F>
__global__ void finalize_bf16_kernel(const float* __restrict__ agg,
                                     const float* __restrict__ h,
                                     const float* __restrict__ dinv,
                                     const float* __restrict__ bias,
                                     __bf16* __restrict__ out, long long n)
{
    long long idx = blockIdx.x * (long long)blockDim.x + threadIdx.x;
    if (idx >= n * F) return;
    long long i = idx / F;
    int f = (int)(idx - i * F);
    float di = dinv[i];
    float v = agg[idx] + h[idx] * di * di + bias[f];
    out[idx] = (__bf16)fmaxf(v, 0.0f);
}

template<int F>
__global__ void finalize_f32_kernel(const float* __restrict__ agg,
                                    const float* __restrict__ h,
                                    const float* __restrict__ dinv,
                                    const float* __restrict__ bias,
                                    float* __restrict__ out, long long n)
{
    long long idx = blockIdx.x * (long long)blockDim.x + threadIdx.x;
    if (idx >= n * F) return;
    long long i = idx / F;
    int f = (int)(idx - i * F);
    float di = dinv[i];
    out[idx] = fmaxf(agg[idx] + h[idx] * di * di + bias[f], 0.0f);
}

// ---------------------------------------------------------------------------
// Host launch
// ---------------------------------------------------------------------------
extern "C" void kernel_launch(void* const* d_in, const int* in_sizes, int n_in,
                              void* d_out, int out_size, void* d_ws, size_t ws_size,
                              hipStream_t stream) {
    const float*     x  = (const float*)d_in[0];
    const long long* ei = (const long long*)d_in[1];  // [2, E] int64
    const float*     ew = (const float*)d_in[2];
    const float*     W1 = (const float*)d_in[3];      // [128, 128]
    const float*     b1 = (const float*)d_in[4];
    const float*     W2 = (const float*)d_in[5];      // [128, 64]
    const float*     b2 = (const float*)d_in[6];

    const long long N = in_sizes[0] / 128;            // 100000
    const long long E = in_sizes[1] / 2;              // 1600000
    constexpr int F1 = 128;                           // 2*NHID
    constexpr int F2 = 64;                            // NHID

    // Workspace layout (all accumulators re-zeroed every call)
    char* p = (char*)d_ws;
    auto take = [&](size_t bytes) -> char* {
        char* r = p; p += (bytes + 255) & ~size_t(255); return r;
    };
    float*  dinv = (float*)take((size_t)N * 4);
    float*  nrm  = (float*)take((size_t)E * 4);
    __bf16* xb   = (__bf16*)take((size_t)N * 128 * 2);
    __bf16* w1t  = (__bf16*)take(128 * 128 * 2);
    __bf16* w2t  = (__bf16*)take(64 * 128 * 2);
    float*  h1   = (float*)take((size_t)N * F1 * 4);
    float*  agg1 = (float*)take((size_t)N * F1 * 4);
    __bf16* o1b  = (__bf16*)take((size_t)N * F1 * 2);
    float*  h2   = (float*)xb;    // alias: xb dead after gemm1 (N*64*4 == N*128*2)
    float*  agg2 = agg1;          // alias: agg1 dead after finalize1

    const int T = 256;
    auto nb = [&](long long total) -> unsigned { return (unsigned)((total + T - 1) / T); };

    // --- degree / normalization ---
    (void)hipMemsetAsync(dinv, 0, (size_t)N * 4, stream);
    deg_kernel<<<nb(E), T, 0, stream>>>(ei, ew, E, dinv);
    rsqrt_kernel<<<nb(N), T, 0, stream>>>(dinv, N);
    norm_kernel<<<nb(E), T, 0, stream>>>(ei, ew, dinv, E, nrm);

    // --- precision conversion (bf16 GEMM path) ---
    conv_bf16_kernel<<<nb(N * 128), T, 0, stream>>>(x, xb, N * 128);
    convw_kernel<<<nb(128 * 128), T, 0, stream>>>(W1, w1t, 128, F1);
    convw_kernel<<<nb(128 * 64),  T, 0, stream>>>(W2, w2t, 128, F2);

    // --- layer 1: h1 = x @ W1 ; agg ; relu ---
    const unsigned g1 = (unsigned)((N / 16 + 7) / 8);
    gemm_bf16_wmma<F1><<<g1, 256, 0, stream>>>(xb, w1t, h1, (int)N);
    (void)hipMemsetAsync(agg1, 0, (size_t)N * F1 * 4, stream);
    scatter_kernel<F1><<<nb(E * (F1 / 4)), T, 0, stream>>>(h1, ei, E, nrm, agg1);
    finalize_bf16_kernel<F1><<<nb(N * F1), T, 0, stream>>>(agg1, h1, dinv, b1, o1b, N);

    // --- layer 2: h2 = relu1 @ W2 ; agg ; relu -> d_out ---
    gemm_bf16_wmma<F2><<<g1, 256, 0, stream>>>(o1b, w2t, h2, (int)N);
    (void)hipMemsetAsync(agg2, 0, (size_t)N * F2 * 4, stream);
    scatter_kernel<F2><<<nb(E * (F2 / 4)), T, 0, stream>>>(h2, ei, E, nrm, agg2);
    finalize_f32_kernel<F2><<<nb(N * F2), T, 0, stream>>>(agg2, h2, dinv, b2,
                                                          (float*)d_out, N);
}
